// FuzzyGNN_74706661146720
// MI455X (gfx1250) — matrix operand order repaired and checked by reading, not compiled, not run on hardware
//
#include <hip/hip_runtime.h>

typedef __attribute__((ext_vector_type(16))) _Float16 v16h;
typedef __attribute__((ext_vector_type(8)))  float    v8f;

#define N_NODES 50000
#define N_EDGES 800000
#define DIM_IN  128
#define H_DIM   256
#define N_LAYERS 3

// ---------------- elementwise / graph-prep kernels ----------------

__global__ void k_init_deg(float* __restrict__ deg, int n) {
  int i = blockIdx.x * blockDim.x + threadIdx.x;
  if (i < n) deg[i] = 1.0f;                       // self-loop contributes 1
}

__global__ void k_accum_deg(const int* __restrict__ dst, float* __restrict__ deg, int e) {
  int i = blockIdx.x * blockDim.x + threadIdx.x;
  if (i < e) atomicAdd(&deg[dst[i]], 1.0f);
}

__global__ void k_finalize_dinv(float* __restrict__ deg, int n) {
  int i = blockIdx.x * blockDim.x + threadIdx.x;
  if (i < n) { float d = deg[i]; deg[i] = (d > 0.0f) ? rsqrtf(d) : 0.0f; }
}

__global__ void k_f32_to_f16(const float* __restrict__ in, _Float16* __restrict__ out, int n) {
  int i = blockIdx.x * blockDim.x + threadIdx.x;
  if (i < n) out[i] = (_Float16)in[i];
}

// W [K,N] f32 row-major -> WT [N,K] f16 (so B-fragment loads are contiguous)
__global__ void k_weight_T16(const float* __restrict__ W, _Float16* __restrict__ WT,
                             int K, int N) {
  int i = blockIdx.x * blockDim.x + threadIdx.x;
  if (i < K * N) {
    int k = i / N, n = i - k * N;
    WT[n * K + k] = (_Float16)W[i];
  }
}

// ---------------- WMMA GEMM: out = act(A[M,K] * W[K,N] + bias) ----------------
// A: f16 row-major [M,K].  BT: f16 [N,K] (transposed weights).
// Block = 8 waves; block owns one 64-column group and stages its B slab
// (64 x K f16, <=32KB) in LDS.  Each wave computes a 16x64 output strip with
// 4 accumulators, A double-buffered from global, B fed from ds_load_b128.
// M % 16 == 0, N % 64 == 0, K % 32 == 0, K <= 256.

__global__ __launch_bounds__(256)
void k_gemm_wmma(const _Float16* __restrict__ A,
                 const _Float16* __restrict__ BT,
                 const float* __restrict__ bias,   // may be nullptr
                 float* __restrict__ outF,         // f32 out or nullptr
                 _Float16* __restrict__ outH,      // f16 out or nullptr
                 int M, int N, int K, int relu)
{
  __shared__ _Float16 Bs[64 * 256];                // 32 KB max (K<=256)

  const int ngroups = N >> 6;                      // 64-col groups
  const int ng      = blockIdx.x % ngroups;
  const int mchunk  = blockIdx.x / ngroups;

  const int wave = threadIdx.x >> 5;
  const int lane = threadIdx.x & 31;
  const int hi   = lane >> 4;                      // lane half (0/1)
  const int lr   = lane & 15;

  // ---- cooperative fill of B slab: rows [ng*64, ng*64+64) x K halves
  {
    const uint4* srcp = (const uint4*)(BT + (size_t)(ng << 6) * K);
    uint4*       dstp = (uint4*)Bs;
    const int n16 = (K << 6) >> 3;                 // 64*K halves / 8 per uint4
    for (int i = threadIdx.x; i < n16; i += 256) dstp[i] = srcp[i];
  }
  __syncthreads();

  const int mt = mchunk * 8 + wave;
  if (mt >= (M >> 4)) return;                      // wave-uniform: EXEC stays all-1
  const int row = (mt << 4) + lr;                  // A row owned by this lane

  const _Float16* arow = A + (size_t)row * K + (hi << 3);
  const int kSteps = K >> 5;

  // A fragment = dwords {0..3, 8..11} from (row, s*32 + hi*8) -> bytes {0..15, 32..47}
  union AU { v16h v; uint4 q[2]; };
  AU a_cur, a_nxt;
  {
    const uint4* ap = (const uint4*)(arow);
    a_cur.q[0] = ap[0];
    a_cur.q[1] = ap[2];
  }

  v8f acc[4] = {};

  for (int s = 0; s < kSteps; ++s) {
    if (s + 1 < kSteps) {                          // prefetch next A fragment
      const uint4* ap = (const uint4*)(arow + ((s + 1) << 5));
      a_nxt.q[0] = ap[0];
      a_nxt.q[1] = ap[2];
    }
    // B fragments from LDS (B 32x16: lanes 0-15 K=0..7,16..23, lanes 16-31 K=8..15,24..31)
    #pragma unroll
    for (int t = 0; t < 4; ++t) {
      union { v16h v; uint4 q[2]; } b;
      const uint4* bp =
          (const uint4*)(Bs + (size_t)((t << 4) + lr) * K + (s << 5) + (hi << 3));
      b.q[0] = bp[0];
      b.q[1] = bp[2];
      acc[t] = __builtin_amdgcn_wmma_f32_16x16x32_f16(
          false, a_cur.v, false, b.v, (short)0, acc[t], false, false);
    }
    a_cur = a_nxt;
  }

  // ---- epilogue: C/D layout element v -> (M = mt*16 + 8*hi + v, N = col)
  #pragma unroll
  for (int t = 0; t < 4; ++t) {
    const int col = (ng << 6) + (t << 4) + lr;
    const float bv = bias ? bias[col] : 0.0f;
    #pragma unroll
    for (int v = 0; v < 8; ++v) {
      const int r = (mt << 4) + (hi << 3) + v;
      float val = acc[t][v] + bv;
      if (relu) val = fmaxf(val, 0.0f);
      if (outF) outF[(size_t)r * N + col] = val;
      if (outH) outH[(size_t)r * N + col] = (_Float16)val;
    }
  }
}

// ---------------- aggregation kernels ----------------

// agg[i,:] = m16[i,:] * dinv[i]^2   (self-loop term; also zero-initializes agg)
__global__ void k_self_init(const _Float16* __restrict__ m16,
                            const float* __restrict__ dinv,
                            float* __restrict__ agg, int n)
{
  int idx = blockIdx.x * blockDim.x + threadIdx.x;   // over n * 128 half-pairs
  if (idx >= n * (H_DIM / 2)) return;
  int node = idx >> 7;
  int cp   = idx & 127;
  float nrm = dinv[node]; nrm *= nrm;
  union { unsigned int u; _Float16 h[2]; } w;
  w.u = *(const unsigned int*)(m16 + (size_t)node * H_DIM + (cp << 1));
  agg[(size_t)node * H_DIM + (cp << 1)]     = (float)w.h[0] * nrm;
  agg[(size_t)node * H_DIM + (cp << 1) + 1] = (float)w.h[1] * nrm;
}

// agg[dst,:] += m16[src,:] * dinv[src]*dinv[dst]   (2 edges per 256-thread block)
__global__ __launch_bounds__(256)
void k_scatter(const int* __restrict__ src, const int* __restrict__ dst,
               const _Float16* __restrict__ m16, const float* __restrict__ dinv,
               float* __restrict__ agg, int E)
{
  int e = blockIdx.x * 2 + (threadIdx.x >> 7);
  if (e >= E) return;
  int cp = threadIdx.x & 127;                      // channel pair 0..127
  int s = src[e], d = dst[e];
  float nrm = dinv[s] * dinv[d];
  union { unsigned int u; _Float16 h[2]; } w;
  w.u = *(const unsigned int*)(m16 + (size_t)s * H_DIM + (cp << 1));
  atomicAdd(&agg[(size_t)d * H_DIM + (cp << 1)],     (float)w.h[0] * nrm);
  atomicAdd(&agg[(size_t)d * H_DIM + (cp << 1) + 1], (float)w.h[1] * nrm);
}

// h16 = f16(relu(agg + b))
__global__ void k_bias_relu_h16(const float* __restrict__ agg,
                                const float* __restrict__ b,
                                _Float16* __restrict__ h16, int n)
{
  int idx = blockIdx.x * blockDim.x + threadIdx.x;
  if (idx < n * H_DIM) {
    int c = idx & (H_DIM - 1);
    h16[idx] = (_Float16)fmaxf(agg[idx] + b[c], 0.0f);
  }
}

// ---------------- launcher ----------------

extern "C" void kernel_launch(void* const* d_in, const int* in_sizes, int n_in,
                              void* d_out, int out_size, void* d_ws, size_t ws_size,
                              hipStream_t stream) {
  const float* x     = (const float*)d_in[0];
  const int*   ei    = (const int*)d_in[1];
  const int*   src   = ei;                 // edge_index[0]
  const int*   dst   = ei + N_EDGES;       // edge_index[1]
  // d_in[2] = edge_attr (unused by reference)
  const float* W_enc = (const float*)d_in[3];
  const float* b_enc = (const float*)d_in[4];
  const float* W_g   = (const float*)d_in[5];
  const float* b_g   = (const float*)d_in[6];
  const float* W_out = (const float*)d_in[7];
  const float* b_out = (const float*)d_in[8];
  float* out = (float*)d_out;

  char* ws = (char*)d_ws;
  size_t off = 0;
  auto wsalloc = [&](size_t bytes) -> void* {
    void* p = ws + off;
    off = (off + bytes + 255) & ~(size_t)255;
    return p;
  };
  float*    dinv = (float*)   wsalloc((size_t)N_NODES * 4);
  _Float16* h16  = (_Float16*)wsalloc((size_t)N_NODES * H_DIM * 2);
  _Float16* m16  = (_Float16*)wsalloc((size_t)N_NODES * H_DIM * 2);
  float*    agg  = (float*)   wsalloc((size_t)N_NODES * H_DIM * 4);
  _Float16* WT   = (_Float16*)wsalloc((size_t)H_DIM * H_DIM * 2);

  const int TB = 256;
  auto cdiv = [](int a, int b) { return (a + b - 1) / b; };

  // GCN normalization: dinv = rsqrt(1 + in-degree)
  k_init_deg<<<cdiv(N_NODES, TB), TB, 0, stream>>>(dinv, N_NODES);
  k_accum_deg<<<cdiv(N_EDGES, TB), TB, 0, stream>>>(dst, dinv, N_EDGES);
  k_finalize_dinv<<<cdiv(N_NODES, TB), TB, 0, stream>>>(dinv, N_NODES);

  // GEMM grid: (N/64) column groups x ceil(Mtiles/8) chunks, 8 waves/block
  const int mchunks    = cdiv(N_NODES / 16, 8);           // 391
  const int gemmBlocks = mchunks * (H_DIM / 64);          // 1564

  // Encoder: h = relu(x @ W_enc + b_enc), f16 activations out
  k_f32_to_f16<<<cdiv(N_NODES * DIM_IN, TB), TB, 0, stream>>>(x, m16, N_NODES * DIM_IN);
  k_weight_T16<<<cdiv(DIM_IN * H_DIM, TB), TB, 0, stream>>>(W_enc, WT, DIM_IN, H_DIM);
  k_gemm_wmma<<<gemmBlocks, TB, 0, stream>>>(m16, WT, b_enc, nullptr, h16,
                                             N_NODES, H_DIM, DIM_IN, /*relu=*/1);

  // 3 GCN layers
  for (int l = 0; l < N_LAYERS; ++l) {
    k_weight_T16<<<cdiv(H_DIM * H_DIM, TB), TB, 0, stream>>>(
        W_g + (size_t)l * H_DIM * H_DIM, WT, H_DIM, H_DIM);
    // m = h @ W_g[l]  (no bias/act; f16 messages halve gather bandwidth)
    k_gemm_wmma<<<gemmBlocks, TB, 0, stream>>>(h16, WT, nullptr, nullptr, m16,
                                               N_NODES, H_DIM, H_DIM, /*relu=*/0);
    // agg = self-loop term, then scatter-add over edges
    k_self_init<<<cdiv(N_NODES * (H_DIM / 2), TB), TB, 0, stream>>>(m16, dinv, agg, N_NODES);
    k_scatter<<<cdiv(N_EDGES, 2), TB, 0, stream>>>(src, dst, m16, dinv, agg, N_EDGES);
    // h = relu(agg + b_g[l])
    k_bias_relu_h16<<<cdiv(N_NODES * H_DIM, TB), TB, 0, stream>>>(
        agg, b_g + (size_t)l * H_DIM, h16, N_NODES);
  }

  // Output head: out = h @ W_out + b_out (f32)
  k_weight_T16<<<cdiv(H_DIM * H_DIM, TB), TB, 0, stream>>>(W_out, WT, H_DIM, H_DIM);
  k_gemm_wmma<<<gemmBlocks, TB, 0, stream>>>(h16, WT, b_out, out, nullptr,
                                             N_NODES, H_DIM, H_DIM, /*relu=*/0);
}
